// MLRelMulcoClassifier_68384469287397
// MI455X (gfx1250) — matrix operand order repaired and checked by reading, not compiled, not run on hardware
//
// MI455X (gfx1250) implementation: RGCN-2 + entity heads + InfoNCE.
// Heavy math runs on v_wmma_f32_16x16x32_bf16 (f32 accumulate). RGCN uses the
// edge-bucketed formulation (per-edge transform + mean-scaled scatter), ~13x
// fewer FLOPs than the per-relation dense scan -> near the HBM roofline
// (~380 MB @ 23.3 TB/s ~= 16 us). GEMMs double-buffer LDS tiles (one barrier
// per K-step, loads overlap WMMA) and stage B with packed dword LDS stores.
#include <hip/hip_runtime.h>
#include <math.h>

constexpr int kBS = 64, kS = 256, kD = 768, kNPG = 128, kEPG = 256;
constexpr int kN = kBS * kNPG;      // 8192 nodes
constexpr int kE = kBS * kEPG;      // 16384 edges
constexpr int kR = 40;              // relation types
constexpr int kH = 256;             // head dim
constexpr int kNR = 10;             // num output relations
constexpr int kPADMAX = kE + 16 * kR;   // 17024 padded edge rows
constexpr int kNTMAX  = kPADMAX / 16;   // 1064 edge tiles (upper bound)

// d_out layout (flat, return order): rel_output, rel_logits, cl_loss, text_doc, graph_doc
constexpr int RO_OFF = 0;
constexpr int RL_OFF = kBS * 3 * kD;          // 147456
constexpr int LOSS_OFF = RL_OFF + kBS * kNR;  // 148096
constexpr int TD_OFF = LOSS_OFF + 1;          // 148097
constexpr int GD_OFF = TD_OFF + kBS * kH;     // 164481

typedef __bf16 bf16_t;
typedef bf16_t v2bf  __attribute__((ext_vector_type(2)));
typedef bf16_t v8bf  __attribute__((ext_vector_type(8)));
typedef bf16_t v16bf __attribute__((ext_vector_type(16)));
typedef float  v8f   __attribute__((ext_vector_type(8)));

// ---------------------------------------------------------------- utilities
__global__ void k_fill_i32(int* __restrict__ p, int n, int val) {
    int i = blockIdx.x * blockDim.x + threadIdx.x;
    if (i < n) p[i] = val;
}

__global__ void k_relu(float* __restrict__ p, int n) {
    int i = blockIdx.x * blockDim.x + threadIdx.x;
    if (i < n) p[i] = fmaxf(p[i], 0.0f);
}

// -------------------------------------------------- relation bucket planning
__global__ void k_count(const int* __restrict__ etype, const int* __restrict__ dst,
                        int* __restrict__ relCount, int* __restrict__ cnt) {
    int e = blockIdx.x * blockDim.x + threadIdx.x;
    if (e < kE) {
        int r = etype[e];
        atomicAdd(&relCount[r], 1);
        atomicAdd(&cnt[(size_t)dst[e] * kR + r], 1);
    }
}

__global__ void k_plan(const int* __restrict__ relCount, int* __restrict__ padStart,
                       int* __restrict__ tileRel, int* __restrict__ cursor) {
    // single thread: 40 relations, <= 1064 tile slots
    int pt = 0, tl = 0;
    for (int r = 0; r < kR; ++r) {
        padStart[r] = pt;
        cursor[r] = 0;
        int tiles = (relCount[r] + 15) >> 4;
        for (int t = 0; t < tiles; ++t) tileRel[tl++] = r;
        pt += tiles * 16;
    }
}

__global__ void k_scatter_edges(const int* __restrict__ etype, const int* __restrict__ padStart,
                                int* __restrict__ cursor, int* __restrict__ eperm) {
    int e = blockIdx.x * blockDim.x + threadIdx.x;
    if (e < kE) {
        int r = etype[e];
        int pos = atomicAdd(&cursor[r], 1);
        eperm[padStart[r] + pos] = e;
    }
}

__global__ void k_edge_meta(const int* __restrict__ eperm, const int* __restrict__ srcA,
                            const int* __restrict__ dstA, const int* __restrict__ etype,
                            const int* __restrict__ cnt, int* __restrict__ esrc,
                            int* __restrict__ edst, float* __restrict__ escale) {
    int i = blockIdx.x * blockDim.x + threadIdx.x;
    if (i >= kPADMAX) return;
    int e = eperm[i];
    if (e < 0) { esrc[i] = -1; edst[i] = -1; escale[i] = 0.0f; return; }
    int d = dstA[e], r = etype[e];
    esrc[i] = srcA[e];
    edst[i] = d;
    int c = cnt[(size_t)d * kR + r];
    escale[i] = 1.0f / (float)(c > 0 ? c : 1);
}

// gather padded per-edge source rows as bf16 (zero rows for padding)
__global__ void k_gather(const float* __restrict__ X, const int* __restrict__ esrc,
                         bf16_t* __restrict__ Xg) {
    int i = blockIdx.x;
    int s = esrc[i];
    for (int c = threadIdx.x; c < kD; c += 256) {
        float v = (s >= 0) ? X[(size_t)s * kD + c] : 0.0f;
        Xg[(size_t)i * kD + c] = (bf16_t)v;
    }
}

// ---------------------------------------------------------- GEMM tile loads
// B tile (32 x 128 f32, row-major, leading dim ldb) -> LDS transposed [n][k]
// bf16. Each thread owns a k-pair x 8 n: packs 2 bf16 into one dword LDS store.
__device__ __forceinline__
void load_b_tile(const float* __restrict__ B, int ldb, int k0, int n0,
                 bf16_t* __restrict__ Bs, int tid) {
    int np = tid >> 4;          // 0..15 -> n base np*8
    int kp = tid & 15;          // k = 2kp, 2kp+1
    const float* b0 = B + (size_t)(k0 + 2 * kp) * ldb + n0 + np * 8;
    const float* b1 = b0 + ldb;
    float4 r0a = reinterpret_cast<const float4*>(b0)[0];
    float4 r0b = reinterpret_cast<const float4*>(b0)[1];
    float4 r1a = reinterpret_cast<const float4*>(b1)[0];
    float4 r1b = reinterpret_cast<const float4*>(b1)[1];
    float r0[8] = {r0a.x, r0a.y, r0a.z, r0a.w, r0b.x, r0b.y, r0b.z, r0b.w};
    float r1[8] = {r1a.x, r1a.y, r1a.z, r1a.w, r1b.x, r1b.y, r1b.z, r1b.w};
#pragma unroll
    for (int j = 0; j < 8; ++j) {
        v2bf pv = {(bf16_t)r0[j], (bf16_t)r1[j]};
        *reinterpret_cast<v2bf*>(&Bs[(np * 8 + j) * 32 + 2 * kp]) = pv;
    }
}

// ------------------------------------------------- dense bf16 WMMA GEMM
// C[M,N] = (relu?)(A[M,K] @ B[K,N] + bias). M%64==0, N%128==0, K%32==0.
// 256 threads = 8 waves; block tile 64x128; wave (wm,wn) does 16x64 via 4
// WMMAs. LDS double-buffered: one barrier per K-step, loads overlap WMMA.
__global__ __launch_bounds__(256)
void k_gemm_dense(const float* __restrict__ A, const float* __restrict__ B,
                  const float* __restrict__ bias, float* __restrict__ C,
                  int M, int N, int K, int do_relu) {
    __shared__ alignas(64) bf16_t As[2][64 * 32];   // [row][k]
    __shared__ alignas(64) bf16_t Bs[2][128 * 32];  // transposed: [n][k]
    const int tid = threadIdx.x, lane = tid & 31, wave = tid >> 5;
    const int wm = wave >> 1, wn = wave & 1;
    const int m0 = blockIdx.y * 64, n0 = blockIdx.x * 128;

    auto load_tiles = [&](int k0, int p) {
        int row = tid >> 2, kk = (tid & 3) * 8;
        const float4* ap = reinterpret_cast<const float4*>(A + (size_t)(m0 + row) * K + k0 + kk);
        float4 f0 = ap[0], f1 = ap[1];
        v8bf v;
        v[0] = (bf16_t)f0.x; v[1] = (bf16_t)f0.y; v[2] = (bf16_t)f0.z; v[3] = (bf16_t)f0.w;
        v[4] = (bf16_t)f1.x; v[5] = (bf16_t)f1.y; v[6] = (bf16_t)f1.z; v[7] = (bf16_t)f1.w;
        *reinterpret_cast<v8bf*>(&As[p][row * 32 + kk]) = v;
        load_b_tile(B, N, k0, n0, Bs[p], tid);
        if (k0 + 64 < K)
            __builtin_prefetch(A + (size_t)(m0 + row) * K + k0 + 64 + kk, 0, 1);
    };

    v8f acc[4] = {};
    load_tiles(0, 0);
    int p = 0;
    for (int k0 = 0; k0 < K; k0 += 32) {
        __syncthreads();
        if (k0 + 32 < K) load_tiles(k0 + 32, p ^ 1);
        {
            const int r = lane & 15, kg = lane >> 4;
            // A fragment: lane row = r, K {16kg..16kg+15}
            union { v16bf v; v8bf h[2]; } af;
            af.v = *reinterpret_cast<const v16bf*>(&As[p][(wm * 16 + r) * 32 + kg * 16]);
#pragma unroll
            for (int s = 0; s < 4; ++s) {
                int nloc = wn * 64 + s * 16 + r;
                v16bf bv = *reinterpret_cast<const v16bf*>(&Bs[p][nloc * 32 + kg * 16]);
                acc[s] = __builtin_amdgcn_wmma_f32_16x16x32_bf16(
                    false, af.v, false, bv, (short)0, acc[s], false, false);
            }
        }
        p ^= 1;
    }
    const int r = lane & 15, kg = lane >> 4;
#pragma unroll
    for (int s = 0; s < 4; ++s) {
        int col = n0 + wn * 64 + s * 16 + r;
        float bv = bias ? bias[col] : 0.0f;
#pragma unroll
        for (int v = 0; v < 8; ++v) {
            int row = m0 + wm * 16 + kg * 8 + v;
            float val = acc[s][v] + bv;
            if (do_relu) val = fmaxf(val, 0.0f);
            C[(size_t)row * N + col] = val;
        }
    }
}

// ------------------------------------- edge-tile WMMA GEMM + mean scatter
// tile t (16 padded edge rows, relation tileRel[t]) x 128 cols (blockIdx.y):
// Y = Xg[tile] @ W_r ; Hout[dst] += Y * (1/cnt). 8 waves = 8 n-subtiles.
__global__ __launch_bounds__(256)
void k_gemm_edge(const bf16_t* __restrict__ Xg, const float* __restrict__ Wrel,
                 const int* __restrict__ tileRel, const int* __restrict__ edst,
                 const float* __restrict__ escale, float* __restrict__ Hout) {
    __shared__ alignas(64) bf16_t As[2][16 * 32];
    __shared__ alignas(64) bf16_t Bs[2][128 * 32];  // transposed [n][k]
    const int t = blockIdx.x;
    const int rel = tileRel[t];
    if (rel < 0) return;  // uniform across block
    const float* W = Wrel + (size_t)rel * kD * kD;
    const int n0 = blockIdx.y * 128;
    const int tid = threadIdx.x, lane = tid & 31, wave = tid >> 5;

    auto load_tiles = [&](int k0, int p) {
        // A tile: 16x32 bf16, one dword (2 elems) per thread
        int row = tid >> 4, kk = (tid & 15) * 2;
        *reinterpret_cast<unsigned int*>(&As[p][row * 32 + kk]) =
            *reinterpret_cast<const unsigned int*>(Xg + (size_t)(t * 16 + row) * kD + k0 + kk);
        load_b_tile(W, kD, k0, n0, Bs[p], tid);
    };

    v8f acc = {};
    load_tiles(0, 0);
    int p = 0;
    for (int k0 = 0; k0 < kD; k0 += 32) {
        __syncthreads();
        if (k0 + 32 < kD) load_tiles(k0 + 32, p ^ 1);
        {
            const int r = lane & 15, kg = lane >> 4;
            union { v16bf v; v8bf h[2]; } af;
            af.v = *reinterpret_cast<const v16bf*>(&As[p][r * 32 + kg * 16]);
            int nloc = wave * 16 + r;
            v16bf bv = *reinterpret_cast<const v16bf*>(&Bs[p][nloc * 32 + kg * 16]);
            acc = __builtin_amdgcn_wmma_f32_16x16x32_bf16(
                false, af.v, false, bv, (short)0, acc, false, false);
        }
        p ^= 1;
    }
    const int r = lane & 15, kg = lane >> 4;
    const int col = n0 + wave * 16 + r;
#pragma unroll
    for (int v = 0; v < 8; ++v) {
        int prow = t * 16 + kg * 8 + v;
        int node = edst[prow];
        if (node >= 0)
            atomicAdd(&Hout[(size_t)node * kD + col], acc[v] * escale[prow]);
    }
}

// ---------------------------------------------------------- pooling / misc
__global__ void k_rowmax_seq(const float* __restrict__ seq, float* __restrict__ out) {
    int d = blockIdx.x * 256 + threadIdx.x;
    int b = blockIdx.y;
    if (d >= kD) return;
    float m = -3.4e38f;
    for (int t = 0; t < kS; ++t)
        m = fmaxf(m, seq[((size_t)b * kS + t) * kD + d]);
    out[(size_t)b * kD + d] = m;
}

__global__ void k_rowmax_nodes(const float* __restrict__ h, float* __restrict__ out) {
    // batch == repeat(arange(BS), NPG): graph b owns nodes [b*NPG, (b+1)*NPG)
    int d = blockIdx.x * 256 + threadIdx.x;
    int b = blockIdx.y;
    if (d >= kD) return;
    float m = -3.4e38f;
    for (int n = 0; n < kNPG; ++n)
        m = fmaxf(m, h[((size_t)b * kNPG + n) * kD + d]);
    out[(size_t)b * kD + d] = m;
}

__global__ void k_entity(const float* __restrict__ seq, const float* __restrict__ emask,
                         const float* __restrict__ h, const float* __restrict__ nmask,
                         float* __restrict__ out) {
    int d = blockIdx.x * 256 + threadIdx.x;
    int b = blockIdx.y;
    if (d >= kD) return;
    float s = 0.0f;
    for (int t = 0; t < kS; ++t)
        s += emask[b * kS + t] * seq[((size_t)b * kS + t) * kD + d];
    for (int n = 0; n < kNPG; ++n) {
        int idx = b * kNPG + n;
        s += h[(size_t)idx * kD + d] * nmask[idx];
    }
    out[(size_t)b * kD + d] = s;
}

__global__ void k_relout(const float* __restrict__ pooled, const float* __restrict__ e1h,
                         const float* __restrict__ e2h, float* __restrict__ out) {
    int i = blockIdx.x * 256 + threadIdx.x;
    if (i >= kBS * 3 * kD) return;
    int b = i / (3 * kD), j = i % (3 * kD);
    float v;
    if (j < kD)            v = pooled[b * kD + j];
    else if (j < 2 * kD)   v = e1h[b * kD + (j - kD)];
    else                   v = e2h[b * kD + (j - 2 * kD)];
    out[i] = tanhf(v);
}

__global__ void k_rellogits(const float* __restrict__ ro, const float* __restrict__ W,
                            const float* __restrict__ bias, float* __restrict__ out) {
    int i = blockIdx.x * 256 + threadIdx.x;
    if (i >= kBS * kNR) return;
    int b = i / kNR, j = i % kNR;
    float s = bias[j];
    for (int k = 0; k < 3 * kD; ++k)
        s += ro[(size_t)b * 3 * kD + k] * W[(size_t)k * kNR + j];
    out[i] = s;
}

__global__ void k_rownorm(const float* __restrict__ X, float* __restrict__ nrm) {
    __shared__ float red[256];
    int row = blockIdx.x, t = threadIdx.x;
    float v = X[(size_t)row * kH + t];
    red[t] = v * v;
    __syncthreads();
    for (int s = 128; s > 0; s >>= 1) {
        if (t < s) red[t] += red[t + s];
        __syncthreads();
    }
    if (t == 0) nrm[row] = fmaxf(sqrtf(red[0]), 1e-12f);
}

__global__ void k_sim(const float* __restrict__ A, const float* __restrict__ B,
                      const float* __restrict__ nA, const float* __restrict__ nB,
                      float* __restrict__ Sm) {
    int idx = blockIdx.x * 256 + threadIdx.x;
    if (idx >= kBS * kBS) return;
    int i = idx >> 6, j = idx & 63;
    float s = 0.0f;
    for (int k = 0; k < kH; ++k)
        s += A[(size_t)i * kH + k] * B[(size_t)j * kH + k];
    Sm[idx] = s / (nA[i] * nB[j]) * 10.0f;  // /TEMP, TEMP=0.1
}

__global__ void k_loss(const float* __restrict__ Sm, float* __restrict__ out) {
    __shared__ float red[kBS];
    int t = threadIdx.x;
    // row t: logsumexp - diag  (info_nce(text, graph))
    float m = -3.4e38f;
    for (int j = 0; j < kBS; ++j) m = fmaxf(m, Sm[t * kBS + j]);
    float s = 0.0f;
    for (int j = 0; j < kBS; ++j) s += expf(Sm[t * kBS + j] - m);
    float l1 = m + logf(s) - Sm[t * kBS + t];
    // col t: logsumexp of Sm^T row t  (info_nce(graph, text))
    float m2 = -3.4e38f;
    for (int i = 0; i < kBS; ++i) m2 = fmaxf(m2, Sm[i * kBS + t]);
    float s2 = 0.0f;
    for (int i = 0; i < kBS; ++i) s2 += expf(Sm[i * kBS + t] - m2);
    float l2 = m2 + logf(s2) - Sm[t * kBS + t];
    red[t] = l1 + l2;
    __syncthreads();
    if (t == 0) {
        float acc = 0.0f;
        for (int k = 0; k < kBS; ++k) acc += red[k];
        out[0] = 0.5f * (acc / (float)kBS);
    }
}

// ------------------------------------------------------------------ driver
extern "C" void kernel_launch(void* const* d_in, const int* in_sizes, int n_in,
                              void* d_out, int out_size, void* d_ws, size_t ws_size,
                              hipStream_t stream) {
    const float* seq     = (const float*)d_in[0];
    const float* pooled  = (const float*)d_in[1];
    const float* e1_mask = (const float*)d_in[2];
    const float* e2_mask = (const float*)d_in[3];
    const float* node_x  = (const float*)d_in[4];
    const float* n1_mask = (const float*)d_in[5];
    const float* n2_mask = (const float*)d_in[6];
    const int*   eidx    = (const int*)d_in[7];
    const int*   etype   = (const int*)d_in[8];
    const float* W_rel1  = (const float*)d_in[10];
    const float* W_root1 = (const float*)d_in[11];
    const float* b1      = (const float*)d_in[12];
    const float* W_rel2  = (const float*)d_in[13];
    const float* W_root2 = (const float*)d_in[14];
    const float* b2      = (const float*)d_in[15];
    const float* bert_W1 = (const float*)d_in[16];
    const float* bert_W2 = (const float*)d_in[17];
    const float* gcn_W1  = (const float*)d_in[18];
    const float* gcn_W2  = (const float*)d_in[19];
    const float* rel_W   = (const float*)d_in[20];
    const float* rel_b   = (const float*)d_in[21];
    const int* src = eidx;          // edge_index[0]
    const int* dst = eidx + kE;     // edge_index[1]
    float* out = (float*)d_out;

    // workspace carve-up (deterministic)
    char* wsb = (char*)d_ws;
    size_t off = 0;
    auto alloc = [&](size_t bytes) -> void* {
        void* p = wsb + off;
        off = (off + bytes + 255) & ~(size_t)255;
        return p;
    };
    float*  h1     = (float*)alloc((size_t)kN * kD * 4);
    float*  h2     = (float*)alloc((size_t)kN * kD * 4);
    bf16_t* Xg     = (bf16_t*)alloc((size_t)kPADMAX * kD * 2);
    int*    cnt    = (int*)alloc((size_t)kN * kR * 4);
    int*    eperm  = (int*)alloc((size_t)kPADMAX * 4);
    int*    edst   = (int*)alloc((size_t)kPADMAX * 4);
    int*    esrc   = (int*)alloc((size_t)kPADMAX * 4);
    float*  escale = (float*)alloc((size_t)kPADMAX * 4);
    int*    relCnt = (int*)alloc(kR * 4);
    int*    padSt  = (int*)alloc(kR * 4);
    int*    cursor = (int*)alloc(kR * 4);
    int*    tileR  = (int*)alloc(kNTMAX * 4);
    float*  tmax   = (float*)alloc((size_t)kBS * kD * 4);
    float*  gmax   = (float*)alloc((size_t)kBS * kD * 4);
    float*  e1h    = (float*)alloc((size_t)kBS * kD * 4);
    float*  e2h    = (float*)alloc((size_t)kBS * kD * 4);
    float*  t1     = (float*)alloc((size_t)kBS * kH * 4);
    float*  g1     = (float*)alloc((size_t)kBS * kH * 4);
    float*  nrm    = (float*)alloc(2 * kBS * 4);
    float*  Sm     = (float*)alloc((size_t)kBS * kBS * 4);
    (void)in_sizes; (void)n_in; (void)out_size; (void)ws_size;
    (void)n2_mask;

    auto blocks = [](int n) { return (n + 255) / 256; };

    // --- relation bucket planning (edge structure is per-call input) ---
    k_fill_i32<<<blocks(kN * kR), 256, 0, stream>>>(cnt, kN * kR, 0);
    k_fill_i32<<<1, 64, 0, stream>>>(relCnt, kR, 0);
    k_fill_i32<<<blocks(kNTMAX), 256, 0, stream>>>(tileR, kNTMAX, -1);
    k_fill_i32<<<blocks(kPADMAX), 256, 0, stream>>>(eperm, kPADMAX, -1);
    k_count<<<blocks(kE), 256, 0, stream>>>(etype, dst, relCnt, cnt);
    k_plan<<<1, 1, 0, stream>>>(relCnt, padSt, tileR, cursor);
    k_scatter_edges<<<blocks(kE), 256, 0, stream>>>(etype, padSt, cursor, eperm);
    k_edge_meta<<<blocks(kPADMAX), 256, 0, stream>>>(eperm, src, dst, etype, cnt,
                                                     esrc, edst, escale);

    dim3 gridRoot(kD / 128, kN / 64);
    dim3 gridEdge(kNTMAX, kD / 128);

    // --- RGCN layer 1: h1 = relu(x@Wroot1 + b1 + sum_r mean_r(x)@W1_r) ---
    k_gather<<<kPADMAX, 256, 0, stream>>>(node_x, esrc, Xg);
    k_gemm_dense<<<gridRoot, 256, 0, stream>>>(node_x, W_root1, b1, h1, kN, kD, kD, 0);
    k_gemm_edge<<<gridEdge, 256, 0, stream>>>(Xg, W_rel1, tileR, edst, escale, h1);
    k_relu<<<blocks(kN * kD), 256, 0, stream>>>(h1, kN * kD);

    // --- RGCN layer 2 ---
    k_gather<<<kPADMAX, 256, 0, stream>>>(h1, esrc, Xg);
    k_gemm_dense<<<gridRoot, 256, 0, stream>>>(h1, W_root2, b2, h2, kN, kD, kD, 0);
    k_gemm_edge<<<gridEdge, 256, 0, stream>>>(Xg, W_rel2, tileR, edst, escale, h2);
    k_relu<<<blocks(kN * kD), 256, 0, stream>>>(h2, kN * kD);

    // --- pooling + entity extraction ---
    dim3 gridBD(kD / 256, kBS);
    k_rowmax_seq<<<gridBD, 256, 0, stream>>>(seq, tmax);
    k_rowmax_nodes<<<gridBD, 256, 0, stream>>>(h2, gmax);
    k_entity<<<gridBD, 256, 0, stream>>>(seq, e1_mask, h2, n1_mask, e1h);
    k_entity<<<gridBD, 256, 0, stream>>>(seq, e2_mask, h2, n2_mask, e2h);

    // --- doc heads (WMMA): relu(max @ W1 @ W2) ---
    dim3 gridHead1(kH / 128, kBS / 64);
    k_gemm_dense<<<gridHead1, 256, 0, stream>>>(tmax, bert_W1, nullptr, t1, kBS, kH, kD, 0);
    k_gemm_dense<<<gridHead1, 256, 0, stream>>>(t1, bert_W2, nullptr, out + TD_OFF, kBS, kH, kH, 1);
    k_gemm_dense<<<gridHead1, 256, 0, stream>>>(gmax, gcn_W1, nullptr, g1, kBS, kH, kD, 0);
    k_gemm_dense<<<gridHead1, 256, 0, stream>>>(g1, gcn_W2, nullptr, out + GD_OFF, kBS, kH, kH, 1);

    // --- relation output + logits ---
    k_relout<<<blocks(kBS * 3 * kD), 256, 0, stream>>>(pooled, e1h, e2h, out + RO_OFF);
    k_rellogits<<<blocks(kBS * kNR), 256, 0, stream>>>(out + RO_OFF, rel_W, rel_b, out + RL_OFF);

    // --- InfoNCE: one 64x64 similarity matrix serves both directions ---
    k_rownorm<<<kBS, 256, 0, stream>>>(out + TD_OFF, nrm);
    k_rownorm<<<kBS, 256, 0, stream>>>(out + GD_OFF, nrm + kBS);
    k_sim<<<blocks(kBS * kBS), 256, 0, stream>>>(out + TD_OFF, out + GD_OFF, nrm, nrm + kBS, Sm);
    k_loss<<<1, kBS, 0, stream>>>(Sm, out + LOSS_OFF);
}
